// CaMLoss_35442070126970
// MI455X (gfx1250) — compile-verified, matrix operand order checked by reading, not run on hardware
//
#include <hip/hip_runtime.h>
#include <hip/hip_bf16.h>
#include <stdint.h>

#define BIGV 1.0e6f
#define EPSV 1.0e-12f

typedef __attribute__((ext_vector_type(16))) __bf16 v16bf;
typedef __attribute__((ext_vector_type(8)))  __bf16 v8bf;
typedef __attribute__((ext_vector_type(8)))  float  v8f;

// ---------------------------------------------------------------- utilities

__global__ void k_fill_u32(uint32_t* p, uint32_t v, size_t n) {
    size_t i = (size_t)blockIdx.x * blockDim.x + threadIdx.x;
    size_t stride = (size_t)gridDim.x * blockDim.x;
    for (; i < n; i += stride) p[i] = v;
}

__device__ inline float block_reduce_sum(float v, float* sh) {
    int tid = threadIdx.x;
    sh[tid] = v;
    __syncthreads();
    for (int s = blockDim.x >> 1; s > 0; s >>= 1) {
        if (tid < s) sh[tid] += sh[tid + s];
        __syncthreads();
    }
    float r = sh[0];
    __syncthreads();
    return r;
}

// -------------------------------------------- 1) centroid segment-sum update

__global__ void k_accum_centers(const float* __restrict__ emb,
                                const int* __restrict__ labels,
                                float* __restrict__ sums,
                                float* __restrict__ counts, int D) {
    int b = blockIdx.x;
    int l = labels[b];
    size_t src = (size_t)b * D;
    size_t dst = (size_t)l * D;
    for (int d = threadIdx.x; d < D; d += blockDim.x)
        atomicAdd(&sums[dst + d], emb[src + d]);
    if (threadIdx.x == 0) atomicAdd(&counts[l], 1.0f);
}

// ------------------------------ 2) finalize centers, normalize, store bf16

__global__ void k_center_norm(const float* __restrict__ centers_value,
                              const float* __restrict__ centers_cnt,
                              const float* __restrict__ sums,
                              const float* __restrict__ counts,
                              __bf16* __restrict__ cenN,
                              float* __restrict__ nsq_cen, int D) {
    __shared__ float sh[256];
    int c = blockIdx.x;
    size_t off = (size_t)c * D;
    float cc = counts[c];
    float s = 0.f;
    for (int d = threadIdx.x; d < D; d += blockDim.x) {
        float cnt = centers_cnt[off + d] + cc;
        float val = centers_value[off + d] + sums[off + d];
        float ce  = (cnt > 0.f) ? val / fmaxf(cnt, 1.f) : 0.f;
        s += ce * ce;
    }
    s = block_reduce_sum(s, sh);
    float inv = 1.f / (sqrtf(s) + EPSV);
    if (threadIdx.x == 0) nsq_cen[c] = s * inv * inv;
    for (int d = threadIdx.x; d < D; d += blockDim.x) {
        float cnt = centers_cnt[off + d] + cc;
        float val = centers_value[off + d] + sums[off + d];
        float ce  = (cnt > 0.f) ? val / fmaxf(cnt, 1.f) : 0.f;
        cenN[off + d] = (__bf16)(ce * inv);
    }
}

// --------------------------------------- 3) normalize embeddings, store bf16

__global__ void k_emb_norm(const float* __restrict__ emb,
                           __bf16* __restrict__ embN,
                           float* __restrict__ nsq_emb, int D) {
    __shared__ float sh[256];
    int b = blockIdx.x;
    size_t off = (size_t)b * D;
    float s = 0.f;
    for (int d = threadIdx.x; d < D; d += blockDim.x) {
        float e = emb[off + d];
        s += e * e;
    }
    s = block_reduce_sum(s, sh);
    float inv = 1.f / (sqrtf(s) + EPSV);
    if (threadIdx.x == 0) nsq_emb[b] = s * inv * inv;
    for (int d = threadIdx.x; d < D; d += blockDim.x)
        embN[off + d] = (__bf16)(emb[off + d] * inv);
}

// ----------------- 4) hardest-positive (cheap: ~B/C same-label rows per row)

__global__ void k_hpos(const __bf16* __restrict__ embN,
                       const __bf16* __restrict__ cenN,
                       const int* __restrict__ labels,
                       const float* __restrict__ nsq_emb,
                       const float* __restrict__ nsq_cen,
                       float* __restrict__ hpos, int B, int D) {
    __shared__ float sh[256];
    int b = blockIdx.x;
    int lb = labels[b];
    float nb = nsq_emb[b];
    size_t boff = (size_t)b * D;
    float best = 0.f;
    for (int j = 0; j < B; ++j) {
        if (labels[j] == lb && j != b) {           // uniform across the block
            size_t joff = (size_t)j * D;
            float p = 0.f;
            for (int d = threadIdx.x; d < D; d += blockDim.x)
                p += (float)embN[boff + d] * (float)embN[joff + d];
            float dot = block_reduce_sum(p, sh);
            float dist = sqrtf(fmaxf(nb + nsq_emb[j] - 2.f * dot, EPSV));
            best = fmaxf(best, dist);
        }
    }
    // own-class center distance (the only unmasked pdc positive)
    {
        size_t coff = (size_t)lb * D;
        float p = 0.f;
        for (int d = threadIdx.x; d < D; d += blockDim.x)
            p += (float)embN[boff + d] * (float)cenN[coff + d];
        float dot = block_reduce_sum(p, sh);
        float dist = sqrtf(fmaxf(nb + nsq_cen[lb] - 2.f * dot, EPSV));
        best = fmaxf(best, dist);
    }
    if (threadIdx.x == 0) hpos[b] = best;
}

// ---- 5) WMMA Gram + fused masked row reductions.
//      Block = 8 waves, owns 64 rows. Each wave computes a 64x64 output tile
//      with 4x4 register blocking: per 32-K step, 8 fragments feed 16 WMMAs
//      (32 FLOP/byte of L2 traffic instead of 8).

__global__ void __launch_bounds__(256)
k_gram_reduce(const __bf16* __restrict__ embN,
              const __bf16* __restrict__ cenN,
              const int* __restrict__ labels,
              const float* __restrict__ nsq_emb,
              const float* __restrict__ nsq_cen,
              const float* __restrict__ hpos,
              int* __restrict__ gMin,     // float bits, per row
              int* __restrict__ gMaxPd,   // float bits
              int* __restrict__ gMaxPdc,  // float bits
              int* __restrict__ gCntPd,
              int* __restrict__ gCntPdc,
              int B, int C, int D) {
    __shared__ int   sMin[64], sMaxPd[64], sMaxPdc[64], sCntPd[64], sCntPdc[64];
    __shared__ float sNsqRow[64], sHpos[64];
    __shared__ int   sLabRow[64];

    const int rowBase = blockIdx.x * 64;
    const int tid = threadIdx.x;
    if (tid < 64) {
        sMin[tid]    = 0x7F7FFFFF;   // FLT_MAX bits
        sMaxPd[tid]  = 0;            // 0.0f bits (matches mask*pd semantics)
        sMaxPdc[tid] = 0;
        sCntPd[tid]  = 0;
        sCntPdc[tid] = 0;
        sNsqRow[tid] = nsq_emb[rowBase + tid];
        sHpos[tid]   = hpos[rowBase + tid];
        sLabRow[tid] = labels[rowBase + tid];
    }
    __syncthreads();

    const int wave = tid >> 5;
    const int lane = tid & 31;
    const int totalCols = B + C;                  // B%64==0, C%64==0 assumed
    const int t = blockIdx.y * 8 + wave;          // 64-column chunk index
    const int colBase = t * 64;

    if (colBase < totalCols) {
        const bool isPd = colBase < B;
        const __bf16* colMat = isPd ? embN : cenN;
        const int colOrg = isPd ? colBase : colBase - B;  // index within colMat

        const int ln = lane & 15;   // fragment row-in-tile / column-in-tile
        const int kg = lane >> 4;   // K-group select per ISA 16-bit layout

        size_t rowOff[4], colOff[4];
#pragma unroll
        for (int i = 0; i < 4; ++i) {
            rowOff[i] = (size_t)(rowBase + i * 16 + ln) * D;
            colOff[i] = (size_t)(colOrg  + i * 16 + ln) * D;
        }

        v8f acc[16];
#pragma unroll
        for (int i = 0; i < 16; ++i)
            acc[i] = (v8f){0.f, 0.f, 0.f, 0.f, 0.f, 0.f, 0.f, 0.f};

        for (int k0 = 0; k0 < D; k0 += 32) {
            v16bf A[4], Bf[4];
#pragma unroll
            for (int i = 0; i < 4; ++i) {
                v8bf a0 = *(const v8bf*)(embN + rowOff[i] + k0 + kg * 8);
                v8bf a1 = *(const v8bf*)(embN + rowOff[i] + k0 + 16 + kg * 8);
                v8bf b0 = *(const v8bf*)(colMat + colOff[i] + k0 + kg * 8);
                v8bf b1 = *(const v8bf*)(colMat + colOff[i] + k0 + 16 + kg * 8);
#pragma unroll
                for (int e = 0; e < 8; ++e) {
                    A[i][e] = a0[e];  A[i][e + 8] = a1[e];
                    Bf[i][e] = b0[e]; Bf[i][e + 8] = b1[e];
                }
            }
#pragma unroll
            for (int r = 0; r < 4; ++r)
#pragma unroll
                for (int c = 0; c < 4; ++c)
                    acc[r * 4 + c] = __builtin_amdgcn_wmma_f32_16x16x32_bf16(
                        false, A[r], false, Bf[c], (short)0, acc[r * 4 + c],
                        false, false);
        }

        // epilogue: within a 16x16 sub-tile, lane holds column n = ln,
        // rows m = v + 8*kg (f32 C/D layout)
#pragma unroll
        for (int c = 0; c < 4; ++c) {
            const int colIdx = colOrg + c * 16 + ln;
            const float nsqC = isPd ? nsq_emb[colIdx] : nsq_cen[colIdx];
            const int colLab = isPd ? labels[colIdx] : colIdx;  // pdc: class id
#pragma unroll
            for (int r = 0; r < 4; ++r) {
#pragma unroll
                for (int v = 0; v < 8; ++v) {
                    const int m = r * 16 + v + kg * 8;     // row within block
                    const float dot = acc[r * 4 + c][v];
                    const float dist =
                        sqrtf(fmaxf(sNsqRow[m] + nsqC - 2.f * dot, EPSV));
                    const bool same = (sLabRow[m] == colLab);
                    const bool isNeg = !same;
                    const bool qual = isNeg && (dist > sHpos[m]);
                    const float val = dist + (qual ? 0.f : BIGV);
                    atomicMin(&sMin[m], __float_as_int(val));
                    if (isNeg) {
                        if (isPd) {
                            atomicMax(&sMaxPd[m], __float_as_int(dist));
                            if (qual) atomicAdd(&sCntPd[m], 1);
                        } else {
                            atomicMax(&sMaxPdc[m], __float_as_int(dist));
                            if (qual) atomicAdd(&sCntPdc[m], 1);
                        }
                    }
                }
            }
        }
    }
    __syncthreads();

    if (tid < 64) {
        const int r = rowBase + tid;
        atomicMin(&gMin[r],    sMin[tid]);
        atomicMax(&gMaxPd[r],  sMaxPd[tid]);
        atomicMax(&gMaxPdc[r], sMaxPdc[tid]);
        atomicAdd(&gCntPd[r],  sCntPd[tid]);
        atomicAdd(&gCntPdc[r], sCntPdc[tid]);
    }
}

// ---------------------------------------------------------- 6) final loss

__global__ void k_final(const float* __restrict__ hpos,
                        const int* __restrict__ gMin,
                        const int* __restrict__ gMaxPd,
                        const int* __restrict__ gMaxPdc,
                        const int* __restrict__ gCntPd,
                        const int* __restrict__ gCntPdc,
                        const float* __restrict__ margin_p,
                        float* __restrict__ out, int B) {
    __shared__ float sh[256];
    const float margin = *margin_p;
    float part = 0.f;
    for (int b = threadIdx.x; b < B; b += blockDim.x) {
        float h_out = __int_as_float(gMin[b]);
        float nin   = (gCntPd[b]  == 0) ? __int_as_float(gMaxPd[b])  : BIGV;
        float ninc  = (gCntPdc[b] == 0) ? __int_as_float(gMaxPdc[b]) : BIGV;
        float h_in  = fmaxf(nin, ninc);
        float hneg  = fminf(h_out, h_in);
        float v = hpos[b] - hneg + margin;
        part += (v > 0.f) ? v : 0.f;
    }
    float total = block_reduce_sum(part, sh);
    if (threadIdx.x == 0) out[0] = total / (float)B;
}

// ---------------------------------------------------------------- launcher

extern "C" void kernel_launch(void* const* d_in, const int* in_sizes, int n_in,
                              void* d_out, int out_size, void* d_ws, size_t ws_size,
                              hipStream_t stream) {
    const float* emb           = (const float*)d_in[0];
    const int*   labels        = (const int*)d_in[1];
    const float* centers_value = (const float*)d_in[2];
    const float* centers_cnt   = (const float*)d_in[3];
    const float* margin        = (const float*)d_in[4];
    float* out = (float*)d_out;

    const int B = in_sizes[1];
    const int D = in_sizes[0] / B;
    const int C = in_sizes[2] / D;

    // workspace layout (all sections 16B-aligned given D%16==0, C%4==0)
    char* ws = (char*)d_ws;
    size_t off = 0;
    float*  centers_sum = (float*)(ws + off); off += (size_t)C * D * sizeof(float);
    float*  counts      = (float*)(ws + off); off += (size_t)C * sizeof(float);
    __bf16* embN        = (__bf16*)(ws + off); off += (size_t)B * D * sizeof(__bf16);
    __bf16* cenN        = (__bf16*)(ws + off); off += (size_t)C * D * sizeof(__bf16);
    float*  nsq_emb     = (float*)(ws + off); off += (size_t)B * sizeof(float);
    float*  nsq_cen     = (float*)(ws + off); off += (size_t)C * sizeof(float);
    float*  hposArr     = (float*)(ws + off); off += (size_t)B * sizeof(float);
    int*    gMin        = (int*)(ws + off);   off += (size_t)B * sizeof(int);
    int*    gMaxPd      = (int*)(ws + off);   off += (size_t)B * sizeof(int);
    int*    gMaxPdc     = (int*)(ws + off);   off += (size_t)B * sizeof(int);
    int*    gCntPd      = (int*)(ws + off);   off += (size_t)B * sizeof(int);
    int*    gCntPdc     = (int*)(ws + off);   off += (size_t)B * sizeof(int);

    // init: centroid accumulators to 0, gMin to FLT_MAX bits, max/cnt to 0
    {
        size_t n0 = (size_t)C * D + C;   // centers_sum + counts (contiguous)
        k_fill_u32<<<2048, 256, 0, stream>>>((uint32_t*)centers_sum, 0u, n0);
        k_fill_u32<<<64, 256, 0, stream>>>((uint32_t*)gMin, 0x7F7FFFFFu, (size_t)B);
        k_fill_u32<<<64, 256, 0, stream>>>((uint32_t*)gMaxPd, 0u, (size_t)B * 4);
    }

    k_accum_centers<<<B, 256, 0, stream>>>(emb, labels, centers_sum, counts, D);
    k_center_norm<<<C, 256, 0, stream>>>(centers_value, centers_cnt,
                                         centers_sum, counts, cenN, nsq_cen, D);
    k_emb_norm<<<B, 256, 0, stream>>>(emb, embN, nsq_emb, D);
    k_hpos<<<B, 256, 0, stream>>>(embN, cenN, labels, nsq_emb, nsq_cen,
                                  hposArr, B, D);

    const int colChunks64 = (B + C) / 64;            // per-wave 64-col chunks
    const int chunks = (colChunks64 + 7) / 8;        // 8 waves per block
    dim3 grid(B / 64, chunks);
    k_gram_reduce<<<grid, 256, 0, stream>>>(embN, cenN, labels, nsq_emb, nsq_cen,
                                            hposArr, gMin, gMaxPd, gMaxPdc,
                                            gCntPd, gCntPdc, B, C, D);

    k_final<<<1, 256, 0, stream>>>(hposArr, gMin, gMaxPd, gMaxPdc,
                                   gCntPd, gCntPdc, margin, out, B);
}
